// GATBlock_25185688224039
// MI455X (gfx1250) — compile-verified, matrix operand order checked by reading, not run on hardware
//
#include <hip/hip_runtime.h>
#include <hip/hip_bf16.h>
#include <math.h>

// ---------------------------------------------------------------------------
// GATv2 block for MI455X (gfx1250, wave32).
//  - Dense GEMMs (x@W_l.T, x@W_r.T, x@W_proj.T) via v_wmma_f32_16x16x32_f16.
//  - Edge phases are memory-bound scatter/gather; x_l/x_r (51MB each) live in
//    the 192MB L2, so gathers are L2-resident. Segment softmax via float
//    atomic max/add.
// ---------------------------------------------------------------------------

typedef __attribute__((ext_vector_type(16))) _Float16 v16h;
typedef __attribute__((ext_vector_type(8)))  float    v8f;

#define GAT_IN   128
#define GAT_H    8
#define GAT_C    32
#define GAT_HC   256
#define NEG_SLOPE 0.2f

// ----- float atomic max via int/uint ordering trick (init to -FLT_MAX) -----
__device__ inline void atomicMaxF32(float* addr, float val) {
    if (val >= 0.0f) atomicMax((int*)addr, __float_as_int(val));
    else             atomicMin((unsigned int*)addr, (unsigned int)__float_as_int(val));
}

// ----- workspace init: mean_sum=0, amax=-FLT_MAX, denom=0, accum=0 ---------
__global__ void gat_init(float* mean_sum, float* amax, float* denom,
                         float* accum, int N) {
    int i = blockIdx.x * blockDim.x + threadIdx.x;
    if (i == 0) mean_sum[0] = 0.0f;
    int nh = N * GAT_H;
    if (i < nh) { amax[i] = -3.402823466e38f; denom[i] = 0.0f; }
    int total = N * GAT_HC;
    int stride = gridDim.x * blockDim.x;
    for (int j = i; j < total; j += stride) accum[j] = 0.0f;
}

// ----- mean of edge_attr (E x 1): wave reduce + atomicAdd ------------------
__global__ void gat_edge_mean(const float* __restrict__ ea, float* mean_sum, int E) {
    int i = blockIdx.x * blockDim.x + threadIdx.x;
    float v = (i < E) ? ea[i] : 0.0f;
    for (int off = 16; off; off >>= 1) v += __shfl_down(v, off, 32);
    if ((threadIdx.x & 31) == 0) atomicAdd(mean_sum, v);
}

// ----- WMMA GEMM: O = X @ W^T --------------------------------------------
// One wave per 16x16 output tile. K=128 -> 4 x v_wmma_f32_16x16x32_f16.
// A fragment (16x32 f16, ISA 7.12.2): lane m = lane&15, kb = (lane>>4)*8,
//   pairs hold K = kb..kb+7 and kb+16..kb+23. B mirrors A for O = X*W^T
//   (both X and W row-major with contiguous K).
// blockIdx.z: 0 -> W_l->x_l, 1 -> W_r->x_r, 2 -> W_proj->proj (2 col tiles).
__global__ __launch_bounds__(32)
void gat_gemm_wmma(const float* __restrict__ X,
                   const float* __restrict__ Wl,
                   const float* __restrict__ Wr,
                   const float* __restrict__ Wp,
                   float* __restrict__ xl,
                   float* __restrict__ xr,
                   float* __restrict__ proj) {
    int mt = blockIdx.x, nt = blockIdx.y, z = blockIdx.z;
    const float* W; float* O; int ldo;
    if      (z == 0) { W = Wl; O = xl;   ldo = GAT_HC; }
    else if (z == 1) { W = Wr; O = xr;   ldo = GAT_HC; }
    else             { if (nt >= 2) return; W = Wp; O = proj; ldo = GAT_C; }

    int lane = threadIdx.x;
    int m    = lane & 15;           // A row / B col within tile
    int kb   = (lane >> 4) * 8;     // per-lane K base
    int row0 = mt * 16, col0 = nt * 16;

    const float* arow = X + (size_t)(row0 + m) * GAT_IN + kb;
    const float* brow = W + (size_t)(col0 + m) * GAT_IN + kb;

    v8f acc = {};
    #pragma unroll
    for (int k0 = 0; k0 < GAT_IN; k0 += 32) {
        v16h a, b;
        #pragma unroll
        for (int i = 0; i < 8; ++i) {
            a[i]     = (_Float16)arow[k0 + i];
            a[i + 8] = (_Float16)arow[k0 + 16 + i];
            b[i]     = (_Float16)brow[k0 + i];
            b[i + 8] = (_Float16)brow[k0 + 16 + i];
        }
        acc = __builtin_amdgcn_wmma_f32_16x16x32_f16(
                  /*neg_a=*/false, a, /*neg_b=*/false, b,
                  /*c_mod=*/(short)0, acc, /*reuse_a=*/false, /*reuse_b=*/false);
    }
    // D layout: VGPR r -> M = r (lanes 0-15) / 8+r (lanes 16-31), N = lane&15
    int n = lane & 15, mb = (lane >> 4) * 8;
    #pragma unroll
    for (int r = 0; r < 8; ++r)
        O[(size_t)(row0 + mb + r) * ldo + col0 + n] = acc[r];
}

// ----- phase 1: attention logits + segment max -----------------------------
// one 256-thread block per edge; wave (32 lanes) per head.
__global__ __launch_bounds__(256)
void gat_logits(const float* __restrict__ xl, const float* __restrict__ xr,
                const int* __restrict__ edge_index,
                const float* __restrict__ edge_attr,
                const float* __restrict__ We, const float* __restrict__ att,
                const float* __restrict__ mean_sum,
                float* __restrict__ alpha, float* __restrict__ amax, int E) {
    int e = blockIdx.x, t = threadIdx.x;
    int h = t >> 5, c = t & 31;
    int src, dst; float ea;
    if (e < E) { src = edge_index[e]; dst = edge_index[E + e]; ea = edge_attr[e]; }
    else       { src = dst = e - E;   ea = mean_sum[0] / (float)E; }

    float m = xl[(size_t)src * GAT_HC + t] + xr[(size_t)dst * GAT_HC + t]
            + ea * We[t];
    m = (m > 0.0f) ? m : NEG_SLOPE * m;              // leaky_relu
    float v = m * att[t];
    for (int off = 16; off; off >>= 1) v += __shfl_down(v, off, 32);
    if (c == 0) {
        alpha[(size_t)e * GAT_H + h] = v;            // raw logit
        atomicMaxF32(&amax[(size_t)dst * GAT_H + h], v);
    }
}

// ----- phase 2: exp(logit - max) + segment sum -----------------------------
__global__ void gat_exp(const int* __restrict__ edge_index,
                        const float* __restrict__ amax,
                        float* __restrict__ alpha, float* __restrict__ denom,
                        int E, int EN) {
    int i = blockIdx.x * blockDim.x + threadIdx.x;
    if (i >= EN * GAT_H) return;
    int e = i >> 3, h = i & 7;
    int dst = (e < E) ? edge_index[E + e] : (e - E);
    float ex = __expf(alpha[i] - amax[(size_t)dst * GAT_H + h]);
    alpha[i] = ex;                                   // store numerator
    atomicAdd(&denom[(size_t)dst * GAT_H + h], ex);
}

// ----- phase 3: normalize + weighted scatter aggregation -------------------
__global__ __launch_bounds__(256)
void gat_aggregate(const float* __restrict__ xl,
                   const int* __restrict__ edge_index,
                   const float* __restrict__ denom,
                   float* __restrict__ alpha, float* __restrict__ accum, int E) {
    int e = blockIdx.x, t = threadIdx.x;
    int h = t >> 5, c = t & 31;
    int src, dst;
    if (e < E) { src = edge_index[e]; dst = edge_index[E + e]; }
    else       { src = dst = e - E; }
    float al = alpha[(size_t)e * GAT_H + h] / denom[(size_t)dst * GAT_H + h];
    if (c == 0) alpha[(size_t)e * GAT_H + h] = al;   // final alpha output
    atomicAdd(&accum[(size_t)dst * GAT_HC + t],
              xl[(size_t)src * GAT_HC + t] * al);
}

// ----- phase 4: head mean + bias + relu + projected residual ---------------
__global__ void gat_finalize(const float* __restrict__ accum,
                             const float* __restrict__ proj,
                             const float* __restrict__ bias,
                             float* __restrict__ xnew, int N) {
    int i = blockIdx.x * blockDim.x + threadIdx.x;
    if (i >= N * GAT_C) return;
    int n = i >> 5, c = i & 31;
    float s = 0.0f;
    #pragma unroll
    for (int h = 0; h < GAT_H; ++h) s += accum[(size_t)n * GAT_HC + h * GAT_C + c];
    s = s * (1.0f / GAT_H) + bias[c];
    s = fmaxf(s, 0.0f);
    xnew[i] = s + proj[i];
}

extern "C" void kernel_launch(void* const* d_in, const int* in_sizes, int n_in,
                              void* d_out, int out_size, void* d_ws, size_t ws_size,
                              hipStream_t stream) {
    const float* x          = (const float*)d_in[0];
    const int*   edge_index = (const int*)  d_in[1];
    const float* edge_attr  = (const float*)d_in[2];
    /* d_in[3] = batch (unused) */
    const float* W_l        = (const float*)d_in[4];
    const float* W_r        = (const float*)d_in[5];
    const float* W_e        = (const float*)d_in[6];
    const float* att        = (const float*)d_in[7];
    const float* bias       = (const float*)d_in[8];
    const float* W_proj     = (const float*)d_in[9];

    const int N  = in_sizes[0] / GAT_IN;   // 50000 (divisible by 16)
    const int E  = in_sizes[1] / 2;        // 400000
    const int EN = E + N;                  // edges incl. self loops

    float* out   = (float*)d_out;
    float* xnew  = out;                    // [N,32]
    float* alpha = out + (size_t)N * GAT_C; // [E+N,8]

    // workspace layout (floats)
    float* ws       = (float*)d_ws;
    float* mean_sum = ws;                                   // 1 (+pad 16)
    float* xl       = ws + 16;                              // N*256
    float* xr       = xl + (size_t)N * GAT_HC;              // N*256
    float* proj     = xr + (size_t)N * GAT_HC;              // N*32
    float* amax     = proj + (size_t)N * GAT_C;             // N*8
    float* denom    = amax + (size_t)N * GAT_H;             // N*8
    float* accum    = denom + (size_t)N * GAT_H;            // N*256

    // 1) init accumulators
    {
        int th = N * GAT_H;
        gat_init<<<(th + 255) / 256, 256, 0, stream>>>(mean_sum, amax, denom, accum, N);
    }
    // 2) edge_attr mean
    gat_edge_mean<<<(E + 255) / 256, 256, 0, stream>>>(edge_attr, mean_sum, E);
    // 3) WMMA GEMMs: x_l, x_r, proj
    {
        dim3 grid(N / 16, GAT_HC / 16, 3);
        gat_gemm_wmma<<<grid, 32, 0, stream>>>(x, W_l, W_r, W_proj, xl, xr, proj);
    }
    // 4) logits + segment max
    gat_logits<<<EN, 256, 0, stream>>>(xl, xr, edge_index, edge_attr, W_e, att,
                                       mean_sum, alpha, amax, E);
    // 5) exp + segment sum
    {
        int th = EN * GAT_H;
        gat_exp<<<(th + 255) / 256, 256, 0, stream>>>(edge_index, amax, alpha, denom, E, EN);
    }
    // 6) normalize + weighted scatter
    gat_aggregate<<<EN, 256, 0, stream>>>(xl, edge_index, denom, alpha, accum, E);
    // 7) finalize nodes
    {
        int th = N * GAT_C;
        gat_finalize<<<(th + 255) / 256, 256, 0, stream>>>(accum, proj, bias, xnew, N);
    }
}
